// MultilayerGRU_49194555408584
// MI455X (gfx1250) — matrix operand order repaired and checked by reading, not compiled, not run on hardware
//
#include <hip/hip_runtime.h>
#include <hip/hip_bf16.h>

#define DI static __device__ __forceinline__

constexpr int Bsz = 64, Ssz = 2048, Isz = 128, Hsz = 512, Osz = 128;
constexpr int H3  = 3 * Hsz;
constexpr int NBLK = 64, NTHR = 256;

typedef __attribute__((ext_vector_type(16))) __bf16        v16bf;
typedef __attribute__((ext_vector_type(8)))  float         v8f;
typedef __attribute__((ext_vector_type(8)))  unsigned int  v8u;

DI unsigned short f2bf(float f) {
  unsigned u = __builtin_bit_cast(unsigned, f);
  u += 0x7FFFu + ((u >> 16) & 1u);              // round-to-nearest-even
  return (unsigned short)(u >> 16);
}
DI float sigmoidf_(float x) { return 1.0f / (1.0f + __expf(-x)); }
DI float tanhf_(float x) {
  x = fminf(fmaxf(x, -15.0f), 15.0f);
  float e = __expf(2.0f * x);
  return (e - 1.0f) / (e + 1.0f);
}

// ---- WMMA fragment loaders (CDNA5 16-bit layouts, ISA 7.12.2) -------------
// A (16x32, row-major MxK): lanes 0-15 -> row M=lane, K pairs {0,2,4,6}+{16..22};
// lanes 16-31 -> same rows, K offset +8.
DI v16bf load_afrag(const unsigned short* A, int lda, int k0, int lane) {
  const int row = lane & 15;
  const int kb  = (lane >> 4) << 3;             // 0 or 8
  const unsigned short* p = A + (size_t)row * lda + k0 + kb;
  uint4 q0 = *(const uint4*)(p);                // K = kb+0..7
  uint4 q1 = *(const uint4*)(p + 16);           // K = 16+kb+0..7
  v8u u; u[0]=q0.x; u[1]=q0.y; u[2]=q0.z; u[3]=q0.w;
         u[4]=q1.x; u[5]=q1.y; u[6]=q1.z; u[7]=q1.w;
  return __builtin_bit_cast(v16bf, u);
}
// B (32x16): lane<16 -> column N=lane, K=0..7,16..23; lane>=16 -> N=lane-16, K=8..15,24..31.
// Weights stored torch-style [N][K] row-major, so a column of B = a row of W.
DI v16bf load_bfrag(const unsigned short* W, int ldw, int k0, int lane) {
  const int n  = lane & 15;
  const int kb = (lane >> 4) << 3;              // 0 or 8
  const unsigned short* p = W + (size_t)n * ldw + k0 + kb;
  uint4 q0 = *(const uint4*)(p);
  uint4 q1 = *(const uint4*)(p + 16);
  v8u u; u[0]=q0.x; u[1]=q0.y; u[2]=q0.z; u[3]=q0.w;
         u[4]=q1.x; u[5]=q1.y; u[6]=q1.z; u[7]=q1.w;
  return __builtin_bit_cast(v16bf, u);
}

// One 16x16 fp32 tile accumulated over K (bf16 A row-major, W = [N][K])
DI v8f gemm_tile(const unsigned short* A, int lda,
                 const unsigned short* W, int ldw,
                 int K, v8f c, int lane) {
  for (int k0 = 0; k0 < K; k0 += 32) {
    v16bf a = load_afrag(A, lda, k0, lane);
    v16bf b = load_bfrag(W, ldw, k0, lane);
    c = __builtin_amdgcn_wmma_f32_16x16x32_bf16(false, a, false, b,
                                                (short)0, c, false, false);
  }
  return c;
}

// ---- grid-wide barrier (all blocks co-resident: 64 blocks only) -----------
DI void grid_sync(unsigned* cnt, unsigned* gen, unsigned nblocks) {
  __syncthreads();
  if (threadIdx.x == 0) {
    __threadfence();
    unsigned g = __hip_atomic_load(gen, __ATOMIC_RELAXED, __HIP_MEMORY_SCOPE_AGENT);
    unsigned a = __hip_atomic_fetch_add(cnt, 1u, __ATOMIC_ACQ_REL, __HIP_MEMORY_SCOPE_AGENT);
    if (a == nblocks - 1u) {
      __hip_atomic_store(cnt, 0u, __ATOMIC_RELAXED, __HIP_MEMORY_SCOPE_AGENT);
      __hip_atomic_fetch_add(gen, 1u, __ATOMIC_ACQ_REL, __HIP_MEMORY_SCOPE_AGENT);
    } else {
      while (__hip_atomic_load(gen, __ATOMIC_ACQUIRE, __HIP_MEMORY_SCOPE_AGENT) == g)
        __builtin_amdgcn_s_sleep(1);
    }
  }
  __syncthreads();
}

// ---- init kernels ---------------------------------------------------------
__global__ void cvt_kernel(const float* __restrict__ src, unsigned short* __restrict__ dst, int n) {
  for (int i = blockIdx.x * blockDim.x + threadIdx.x; i < n; i += gridDim.x * blockDim.x)
    dst[i] = f2bf(src[i]);
}

__global__ void init_state_kernel(const float* __restrict__ h0, const float* __restrict__ x,
                                  unsigned short* hb0, unsigned short* hb1,
                                  float* hf0, float* hf1, unsigned short* xb,
                                  unsigned* syncp) {
  const int tid = blockIdx.x * blockDim.x + threadIdx.x;
  const int NT  = gridDim.x * blockDim.x;
  if (tid == 0) { syncp[0] = 0u; syncp[1] = 0u; }
  for (int i = tid; i < Bsz * Hsz; i += NT) {
    const int b = i / Hsz, j = i % Hsz;
    float v0 = h0[((size_t)b * 2 + 0) * Hsz + j];
    float v1 = h0[((size_t)b * 2 + 1) * Hsz + j];
    hf0[i] = v0; hb0[i] = f2bf(v0);
    hf1[i] = v1; hb1[i] = f2bf(v1);
  }
  for (int i = tid; i < Bsz * Isz; i += NT) {
    const int b = i / Isz, ii = i % Isz;
    xb[i] = f2bf(x[(size_t)b * Ssz * Isz + ii]);   // x[:,0,:]
  }
}

// ---- persistent GRU kernel ------------------------------------------------
__global__ __launch_bounds__(NTHR) void gru_persistent(
    const float* __restrict__ x,
    const float* __restrict__ bh0, const float* __restrict__ bh1,
    const float* __restrict__ by,
    const unsigned short* __restrict__ Wx0b, const unsigned short* __restrict__ Wh0b,
    const unsigned short* __restrict__ Wx1b, const unsigned short* __restrict__ Wh1b,
    const unsigned short* __restrict__ Whyb,
    unsigned short* hb0, unsigned short* hb1,
    unsigned short* rh0, unsigned short* rh1,
    unsigned short* xb,
    float* hf0, float* hf1, float* z0, float* z1, float* a1g,
    unsigned* syncp, float* out) {

  const int tid  = (int)(blockIdx.x * blockDim.x + threadIdx.x);
  const int NT   = (int)(gridDim.x * blockDim.x);
  const int wid  = tid >> 5;
  const int lane = (int)(threadIdx.x & 31);
  const int NW   = NT >> 5;
  const int rsel = (lane >> 4) << 3;   // accumulator row offset: 0 or 8
  const int nloc = lane & 15;          // accumulator column within tile
  unsigned* cnt = syncp; unsigned* gen = syncp + 1;

  for (int t = 0; t < Ssz; ++t) {
    // ---- phase 1: layer0 z/r projection  (+ output head for t-1) ----------
    for (int job = wid; job < 256 + 32; job += NW) {
      if (job < 256) {
        const int ct = job & 63, mt = job >> 6;         // gate cols [0,2H)
        v8f c = {};
        c = gemm_tile(xb  + (size_t)mt * 16 * Isz, Isz,
                      Wx0b + (size_t)(ct * 16) * Isz, Isz, Isz, c, lane);
        c = gemm_tile(hb0 + (size_t)mt * 16 * Hsz, Hsz,
                      Wh0b + (size_t)(ct * 16) * Hsz, Hsz, Hsz, c, lane);
        const int col = ct * 16 + nloc;
        const float bias = bh0[col];
#pragma unroll
        for (int v = 0; v < 8; ++v) {
          const int m = mt * 16 + v + rsel;
          const float s = sigmoidf_(c[v] + bias);
          if (col < Hsz) z0[m * Hsz + col] = s;                               // z gate
          else { const int j = col - Hsz;                                     // r gate
                 rh0[m * Hsz + j] = f2bf(s * hf0[m * Hsz + j]); }
        }
      } else if (t > 0) {                                // y_{t-1} = h1@Why^T + by
        const int e = job - 256, nt = e & 7, mt = e >> 3;
        v8f c = {};
        c = gemm_tile(hb1 + (size_t)mt * 16 * Hsz, Hsz,
                      Whyb + (size_t)(nt * 16) * Hsz, Hsz, Hsz, c, lane);
        const int o = nt * 16 + nloc;
        const float bias = by[o];
#pragma unroll
        for (int v = 0; v < 8; ++v) {
          const int m = mt * 16 + v + rsel;
          out[((size_t)m * Ssz + (t - 1)) * Osz + o] = c[v] + bias;
        }
      }
    }
    grid_sync(cnt, gen, gridDim.x);

    // ---- phase 2: layer0 g gate + state combine ----------------------------
    for (int job = wid; job < 128; job += NW) {
      const int jt = job & 31, mt = job >> 5;
      v8f c = {};
      c = gemm_tile(xb  + (size_t)mt * 16 * Isz, Isz,
                    Wx0b + (size_t)(2 * Hsz + jt * 16) * Isz, Isz, Isz, c, lane);
      c = gemm_tile(rh0 + (size_t)mt * 16 * Hsz, Hsz,
                    Wh0b + (size_t)(2 * Hsz + jt * 16) * Hsz, Hsz, Hsz, c, lane);
      const int j = jt * 16 + nloc;
      const float bias = bh0[2 * Hsz + j];
#pragma unroll
      for (int v = 0; v < 8; ++v) {
        const int m = mt * 16 + v + rsel;
        const float g  = tanhf_(c[v] + bias);
        const float zz = z0[m * Hsz + j];
        const float h  = hf0[m * Hsz + j];
        const float hn = zz * h + (1.0f - zz) * g;
        hf0[m * Hsz + j] = hn;
        hb0[m * Hsz + j] = f2bf(hn);
      }
    }
    grid_sync(cnt, gen, gridDim.x);

    // ---- phase 3: xp1 + layer1 z/r; stash a1g ------------------------------
    for (int job = wid; job < 256 + 128; job += NW) {
      if (job < 256) {
        const int ct = job & 63, mt = job >> 6;
        v8f c = {};
        c = gemm_tile(hb0 + (size_t)mt * 16 * Hsz, Hsz,
                      Wx1b + (size_t)(ct * 16) * Hsz, Hsz, Hsz, c, lane);
        c = gemm_tile(hb1 + (size_t)mt * 16 * Hsz, Hsz,
                      Wh1b + (size_t)(ct * 16) * Hsz, Hsz, Hsz, c, lane);
        const int col = ct * 16 + nloc;
        const float bias = bh1[col];
#pragma unroll
        for (int v = 0; v < 8; ++v) {
          const int m = mt * 16 + v + rsel;
          const float s = sigmoidf_(c[v] + bias);
          if (col < Hsz) z1[m * Hsz + col] = s;
          else { const int j = col - Hsz;
                 rh1[m * Hsz + j] = f2bf(s * hf1[m * Hsz + j]); }
        }
      } else {                                           // a1g = h0'@Wx1_g^T
        const int q = job - 256, jt = q & 31, mt = q >> 5;
        v8f c = {};
        c = gemm_tile(hb0 + (size_t)mt * 16 * Hsz, Hsz,
                      Wx1b + (size_t)(2 * Hsz + jt * 16) * Hsz, Hsz, Hsz, c, lane);
        const int j = jt * 16 + nloc;
#pragma unroll
        for (int v = 0; v < 8; ++v)
          a1g[(mt * 16 + v + rsel) * Hsz + j] = c[v];
      }
    }
    grid_sync(cnt, gen, gridDim.x);

    // ---- phase 4: layer1 g gate + combine; prefetch/convert x_{t+1} --------
    for (int job = wid; job < 128; job += NW) {
      const int jt = job & 31, mt = job >> 5;
      v8f c = {};
      c = gemm_tile(rh1 + (size_t)mt * 16 * Hsz, Hsz,
                    Wh1b + (size_t)(2 * Hsz + jt * 16) * Hsz, Hsz, Hsz, c, lane);
      const int j = jt * 16 + nloc;
      const float bias = bh1[2 * Hsz + j];
#pragma unroll
      for (int v = 0; v < 8; ++v) {
        const int m = mt * 16 + v + rsel;
        const float g  = tanhf_(c[v] + a1g[m * Hsz + j] + bias);
        const float zz = z1[m * Hsz + j];
        const float h  = hf1[m * Hsz + j];
        const float hn = zz * h + (1.0f - zz) * g;
        hf1[m * Hsz + j] = hn;
        hb1[m * Hsz + j] = f2bf(hn);
      }
    }
    if (t + 1 < Ssz) {
      for (int idx = tid; idx < Bsz * Isz; idx += NT) {
        const int b = idx >> 7, i = idx & (Isz - 1);
        xb[idx] = f2bf(x[((size_t)b * Ssz + (t + 1)) * Isz + i]);
      }
    }
    grid_sync(cnt, gen, gridDim.x);
  }

  // ---- tail: y_{S-1} and final hidden state --------------------------------
  for (int job = wid; job < 32; job += NW) {
    const int nt = job & 7, mt = job >> 3;
    v8f c = {};
    c = gemm_tile(hb1 + (size_t)mt * 16 * Hsz, Hsz,
                  Whyb + (size_t)(nt * 16) * Hsz, Hsz, Hsz, c, lane);
    const int o = nt * 16 + nloc;
    const float bias = by[o];
#pragma unroll
    for (int v = 0; v < 8; ++v) {
      const int m = mt * 16 + v + rsel;
      out[((size_t)m * Ssz + (Ssz - 1)) * Osz + o] = c[v] + bias;
    }
  }
  const size_t base = (size_t)Bsz * Ssz * Osz;
  for (int idx = tid; idx < Bsz * Hsz; idx += NT) {
    const int b = idx / Hsz, j = idx % Hsz;
    out[base + ((size_t)b * 2 + 0) * Hsz + j] = hf0[idx];
    out[base + ((size_t)b * 2 + 1) * Hsz + j] = hf1[idx];
  }
}

// ---- host launcher --------------------------------------------------------
extern "C" void kernel_launch(void* const* d_in, const int* in_sizes, int n_in,
                              void* d_out, int out_size, void* d_ws, size_t ws_size,
                              hipStream_t stream) {
  (void)in_sizes; (void)n_in; (void)out_size; (void)ws_size;
  const float* x   = (const float*)d_in[0];
  const float* h0  = (const float*)d_in[1];
  const float* Wx0 = (const float*)d_in[2];
  const float* Wh0 = (const float*)d_in[3];
  const float* bh0 = (const float*)d_in[4];
  const float* Wx1 = (const float*)d_in[5];
  const float* Wh1 = (const float*)d_in[6];
  const float* bh1 = (const float*)d_in[7];
  const float* Why = (const float*)d_in[8];
  const float* by  = (const float*)d_in[9];
  float* out = (float*)d_out;

  char* ws = (char*)d_ws;
  size_t off = 0;
  auto alloc = [&](size_t bytes) -> void* {
    void* p = ws + off;
    off = (off + bytes + 255) & ~(size_t)255;
    return p;
  };
  unsigned short* Wx0b = (unsigned short*)alloc((size_t)H3 * Isz * 2);
  unsigned short* Wh0b = (unsigned short*)alloc((size_t)H3 * Hsz * 2);
  unsigned short* Wx1b = (unsigned short*)alloc((size_t)H3 * Hsz * 2);
  unsigned short* Wh1b = (unsigned short*)alloc((size_t)H3 * Hsz * 2);
  unsigned short* Whyb = (unsigned short*)alloc((size_t)Osz * Hsz * 2);
  unsigned short* hb0  = (unsigned short*)alloc((size_t)Bsz * Hsz * 2);
  unsigned short* hb1  = (unsigned short*)alloc((size_t)Bsz * Hsz * 2);
  unsigned short* rh0  = (unsigned short*)alloc((size_t)Bsz * Hsz * 2);
  unsigned short* rh1  = (unsigned short*)alloc((size_t)Bsz * Hsz * 2);
  unsigned short* xb   = (unsigned short*)alloc((size_t)Bsz * Isz * 2);
  float* hf0 = (float*)alloc((size_t)Bsz * Hsz * 4);
  float* hf1 = (float*)alloc((size_t)Bsz * Hsz * 4);
  float* z0  = (float*)alloc((size_t)Bsz * Hsz * 4);
  float* z1  = (float*)alloc((size_t)Bsz * Hsz * 4);
  float* a1g = (float*)alloc((size_t)Bsz * Hsz * 4);
  unsigned* syncp = (unsigned*)alloc(256);

  cvt_kernel<<<512, 256, 0, stream>>>(Wx0, Wx0b, H3 * Isz);
  cvt_kernel<<<512, 256, 0, stream>>>(Wh0, Wh0b, H3 * Hsz);
  cvt_kernel<<<512, 256, 0, stream>>>(Wx1, Wx1b, H3 * Hsz);
  cvt_kernel<<<512, 256, 0, stream>>>(Wh1, Wh1b, H3 * Hsz);
  cvt_kernel<<<512, 256, 0, stream>>>(Why, Whyb, Osz * Hsz);
  init_state_kernel<<<128, 256, 0, stream>>>(h0, x, hb0, hb1, hf0, hf1, xb, syncp);

  gru_persistent<<<NBLK, NTHR, 0, stream>>>(
      x, bh0, bh1, by,
      Wx0b, Wh0b, Wx1b, Wh1b, Whyb,
      hb0, hb1, rh0, rh1, xb,
      hf0, hf1, z0, z1, a1g,
      syncp, out);
}
